// PrunedRNNModel_47691316855453
// MI455X (gfx1250) — compile-verified, matrix operand order checked by reading, not run on hardware
//
#include <hip/hip_runtime.h>
#include <hip/hip_bf16.h>
#include <math.h>

// Problem constants from the reference: B,T,F,L,H,O = 32,12,4,1024,64,1
#define B_ 32
#define T_ 12
#define F_ 4
#define L_ 1024
#define H_ 64

typedef __bf16 bf16;
typedef __attribute__((ext_vector_type(16))) __bf16 v16bf;
typedef __attribute__((ext_vector_type(8)))  float  v8f;

// K index within a 32-wide K tile for fragment slot j (0..15) given lane half.
// Matches CDNA5 16-bit A/B VGPR layout:
//   lanes 0-15 : VGPR0-3 hold K=0..7,  VGPR4-7 hold K=16..23
//   lanes 16-31: VGPR0-3 hold K=8..15, VGPR4-7 hold K=24..31
__device__ __forceinline__ int frag_k_local(int half, int j) {
  return half * 8 + ((j < 8) ? j : (j + 8));
}

// ---------------------------------------------------------------------------
// Prologue: zero h0
__global__ void zero_kernel(float* __restrict__ p, int n) {
  int i = blockIdx.x * blockDim.x + threadIdx.x;
  if (i < n) p[i] = 0.0f;
}

// Prologue: transpose x [B,T,F,L] -> xt [T][L][B][F]  (link-contiguous per step)
__global__ void transpose_x_kernel(const float* __restrict__ x, float* __restrict__ xt) {
  int idx = blockIdx.x * blockDim.x + threadIdx.x;   // dest-linear index
  int f = idx & 3;
  int r = idx >> 2;
  int b = r & 31;
  r >>= 5;
  int l = r & 1023;
  int t = r >> 10;
  xt[idx] = x[((b * T_ + t) * F_ + f) * L_ + l];
}

// Prologue: pack w_rh/w_uh/w_nh [L,64,64] f32 into bf16 WMMA B-fragments.
// wpack layout: [L][gate(3)][frag(8)=kt*4+nt][lane(32)*16+j] -- so the step
// kernel's per-lane fragment load is a contiguous, coalesced 32B read.
__global__ __launch_bounds__(256) void pack_w_kernel(
    const float* __restrict__ w_rh, const float* __restrict__ w_uh,
    const float* __restrict__ w_nh, bf16* __restrict__ wpack) {
  const int l = blockIdx.x;
  for (int d = threadIdx.x; d < 3 * 8 * 512; d += 256) {
    int g   = d / 4096;
    int rem = d - g * 4096;
    int frag = rem >> 9;        // 0..7
    int li   = rem & 511;       // lane*16 + j
    int kt = frag >> 2, nt = frag & 3;
    int lane = li >> 4, j = li & 15;
    int half = lane >> 4;
    int K = kt * 32 + frag_k_local(half, j);     // contraction index (h)
    int N = nt * 16 + (lane & 15);               // output index (k)
    const float* wg = (g == 0) ? w_rh : (g == 1 ? w_uh : w_nh);
    wpack[(size_t)l * 12288 + d] = (bf16)wg[(size_t)l * 4096 + K * 64 + N];
  }
}

// ---------------------------------------------------------------------------
// One GRU time step. One workgroup (8 wave32) per link.
// h layout: [L][B][H] link-major so all per-link traffic is contiguous (L2-resident).
__global__ __launch_bounds__(256) void gru_step_kernel(
    const float* __restrict__ h_prev,      // [L][B][H]
    float*       __restrict__ h_next,      // [L][B][H]
    const float* __restrict__ xt,          // [L][B][F] (this step's slice)
    const float* __restrict__ att,         // [L][L] (tridiagonal values read at runtime)
    const bf16*  __restrict__ wpack,       // packed gate weights
    const float* __restrict__ w_ri, const float* __restrict__ b_ri,
    const float* __restrict__ w_ui, const float* __restrict__ b_ui,
    const float* __restrict__ w_ni, const float* __restrict__ b_ni,
    const float* __restrict__ b_rh, const float* __restrict__ b_uh,
    const float* __restrict__ b_nh) {
  __shared__ bf16 hatt[B_ * H_];           // A matrix, [b=32][h=64], bf16

  const int l   = blockIdx.x;
  const int tid = threadIdx.x;

  // Tridiagonal attention coefficients for row l (block-uniform, no divergence).
  const float a_d  = att[(size_t)l * L_ + l];
  const float a_lm = (l > 0)      ? att[(size_t)l * L_ + l - 1] : 0.0f;
  const float a_rp = (l < L_ - 1) ? att[(size_t)l * L_ + l + 1] : 0.0f;
  const float* hc = h_prev + (size_t)l * (B_ * H_);
  const float* hl = h_prev + (size_t)(l > 0 ? l - 1 : l) * (B_ * H_);
  const float* hr = h_prev + (size_t)(l < L_ - 1 ? l + 1 : l) * (B_ * H_);

  // Cooperative hatt stencil: hatt[b,h] = a_d*h[l] + a_lm*h[l-1] + a_rp*h[l+1]
  #pragma unroll
  for (int i = tid; i < B_ * H_; i += 256)
    hatt[i] = (bf16)(a_d * hc[i] + a_lm * hl[i] + a_rp * hr[i]);
  __syncthreads();

  const int wave = tid >> 5, lane = tid & 31;
  const int mt = wave >> 2, nt = wave & 3;       // 2 M-tiles x 4 N-tiles
  const int half = lane >> 4;
  const int mrow = mt * 16 + (lane & 15);

  // A fragments (one per K-step), gathered from LDS per the 16-bit A layout:
  // two contiguous 8-element bf16 runs per lane.
  v16bf afrag[2];
  #pragma unroll
  for (int kt = 0; kt < 2; ++kt) {
    const bf16* p = &hatt[mrow * H_ + kt * 32 + half * 8];
    #pragma unroll
    for (int j = 0; j < 8; ++j) { afrag[kt][j] = p[j]; afrag[kt][j + 8] = p[j + 16]; }
  }

  // 3 gate GEMMs (r,u,n), K=64 -> 2 WMMAs each, B-fragments coalesced from L2.
  v8f acc[3] = {v8f{}, v8f{}, v8f{}};
  const bf16* wl = wpack + (size_t)l * 12288;
  #pragma unroll
  for (int g = 0; g < 3; ++g) {
    #pragma unroll
    for (int kt = 0; kt < 2; ++kt) {
      v16bf bfrag = *reinterpret_cast<const v16bf*>(wl + g * 4096 + (kt * 4 + nt) * 512 + lane * 16);
      acc[g] = __builtin_amdgcn_wmma_f32_16x16x32_bf16(
          /*neg_a=*/false, afrag[kt], /*neg_b=*/false, bfrag,
          /*c_mod=*/(short)0, acc[g], /*reuse_a=*/false, /*reuse_b=*/false);
    }
  }

  // Epilogue per C/D layout: N (=k) fixed per lane; M (=b) varies over the 8 VGPRs.
  const int k = nt * 16 + (lane & 15);
  const float brh = b_rh[k * L_ + l], bri = b_ri[k * L_ + l];
  const float buh = b_uh[k * L_ + l], bui = b_ui[k * L_ + l];
  const float bnh = b_nh[k * L_ + l], bni = b_ni[k * L_ + l];
  float wri[F_], wui[F_], wni[F_];
  #pragma unroll
  for (int f = 0; f < F_; ++f) {
    wri[f] = w_ri[(l * F_ + f) * H_ + k];
    wui[f] = w_ui[(l * F_ + f) * H_ + k];
    wni[f] = w_ni[(l * F_ + f) * H_ + k];
  }
  const float* xl = xt + (size_t)l * (B_ * F_);
  #pragma unroll
  for (int vi = 0; vi < 8; ++vi) {
    int b = mt * 16 + ((lane < 16) ? vi : vi + 8);
    float xr = 0.f, xz = 0.f, xn = 0.f;
    #pragma unroll
    for (int f = 0; f < F_; ++f) {
      float xv = xl[b * F_ + f];
      xr += xv * wri[f]; xz += xv * wui[f]; xn += xv * wni[f];
    }
    float hcv = hc[b * H_ + k];
    float r = 1.0f / (1.0f + __expf(-(acc[0][vi] + brh + xr + bri)));
    float z = 1.0f / (1.0f + __expf(-(acc[1][vi] + buh + xz + bui)));
    float n = tanhf(xn + bni + r * (acc[2][vi] + bnh));
    h_next[(size_t)l * (B_ * H_) + b * H_ + k] = (1.0f - z) * n + z * hcv;
  }
}

// ---------------------------------------------------------------------------
// Output head: out[b,0,l] = sum_k hT[l][b][k]*w_fc[l,k,0] + b_fc[0,l]
__global__ void head_kernel(const float* __restrict__ hT, const float* __restrict__ w_fc,
                            const float* __restrict__ b_fc, float* __restrict__ out) {
  int idx = blockIdx.x * blockDim.x + threadIdx.x;   // B*L threads
  int l = idx & (L_ - 1);
  int b = idx >> 10;
  const float* h = hT + (size_t)l * (B_ * H_) + b * H_;
  const float* w = w_fc + (size_t)l * H_;            // O == 1
  float s = 0.0f;
  #pragma unroll
  for (int kk = 0; kk < H_; ++kk) s += h[kk] * w[kk];
  out[b * L_ + l] = s + b_fc[l];
}

// ---------------------------------------------------------------------------
extern "C" void kernel_launch(void* const* d_in, const int* in_sizes, int n_in,
                              void* d_out, int out_size, void* d_ws, size_t ws_size,
                              hipStream_t stream) {
  (void)in_sizes; (void)n_in; (void)out_size; (void)ws_size;
  const float* x    = (const float*)d_in[0];
  const float* att  = (const float*)d_in[1];
  const float* w_rh = (const float*)d_in[2];
  const float* b_rh = (const float*)d_in[3];
  const float* w_ri = (const float*)d_in[4];
  const float* b_ri = (const float*)d_in[5];
  const float* w_uh = (const float*)d_in[6];
  const float* b_uh = (const float*)d_in[7];
  const float* w_ui = (const float*)d_in[8];
  const float* b_ui = (const float*)d_in[9];
  const float* w_nh = (const float*)d_in[10];
  const float* b_nh = (const float*)d_in[11];
  const float* w_ni = (const float*)d_in[12];
  const float* b_ni = (const float*)d_in[13];
  const float* w_fc = (const float*)d_in[14];
  const float* b_fc = (const float*)d_in[15];
  float* out = (float*)d_out;

  // Workspace layout (total 46 MB):
  //   [0,8M)    h ping  [L][B][H] f32
  //   [8M,16M)  h pong
  //   [16M,22M) xt [T][L][B][F] f32
  //   [22M,46M) packed bf16 gate weights [L][3][8][512]
  char* ws = (char*)d_ws;
  float* h0 = (float*)ws;
  float* h1 = (float*)(ws + (size_t)8  * 1024 * 1024);
  float* xt = (float*)(ws + (size_t)16 * 1024 * 1024);
  bf16*  wp = (bf16*) (ws + (size_t)22 * 1024 * 1024);

  zero_kernel<<<(L_ * B_ * H_ + 255) / 256, 256, 0, stream>>>(h0, L_ * B_ * H_);
  transpose_x_kernel<<<(B_ * T_ * F_ * L_) / 256, 256, 0, stream>>>(x, xt);
  pack_w_kernel<<<L_, 256, 0, stream>>>(w_rh, w_uh, w_nh, wp);

  float* hin = h0; float* hout = h1;
  for (int t = 0; t < T_; ++t) {
    gru_step_kernel<<<L_, 256, 0, stream>>>(hin, hout, xt + (size_t)t * L_ * B_ * F_,
        att, wp, w_ri, b_ri, w_ui, b_ui, w_ni, b_ni, b_rh, b_uh, b_nh);
    float* tmp = hin; hin = hout; hout = tmp;
  }
  head_kernel<<<(B_ * L_) / 256, 256, 0, stream>>>(hin, w_fc, b_fc, out);
}
